// LocalizeModule_87101936763140
// MI455X (gfx1250) — matrix operand order repaired and checked by reading, not compiled, not run on hardware
//
#include <hip/hip_runtime.h>
#include <hip/hip_bf16.h>
#include <cstdint>
#include <cstddef>

typedef __attribute__((ext_vector_type(16))) __bf16 v16bf;
typedef __attribute__((ext_vector_type(8)))  float  v8f;

// ---------- helpers ----------
__device__ inline unsigned short f32_to_bf16_rne(float x) {
    uint32_t u = __builtin_bit_cast(uint32_t, x);
    uint32_t r = u + 0x7FFFu + ((u >> 16) & 1u);
    return (unsigned short)(r >> 16);
}
__device__ inline float bf16_to_f32(unsigned short h) {
    uint32_t u = ((uint32_t)h) << 16;
    return __builtin_bit_cast(float, u);
}
__device__ inline v16bf pack_frag(uint4 a, uint4 b) {
    union { uint4 q[2]; v16bf v; } u;
    u.q[0] = a; u.q[1] = b;
    return u.v;
}

// CDNA5 async global->LDS copy, 16B per lane, tracked by ASYNCcnt.
// VDST = per-lane LDS byte offset, VADDR = per-lane 64-bit global address.
__device__ inline void async_b128(uint32_t lds_off, uint64_t gaddr) {
    asm volatile("global_load_async_to_lds_b128 %0, %1, off"
                 :: "v"(lds_off), "v"(gaddr) : "memory");
}
__device__ inline void wait_async_le16() {
    asm volatile("s_wait_asynccnt 16" ::: "memory");
}
__device__ inline void wait_async_0() {
    asm volatile("s_wait_asynccnt 0" ::: "memory");
}

// ---------- fp32 -> bf16 conversion (4-wide) ----------
struct us4 { unsigned short x, y, z, w; };

__global__ void convert_f32_to_bf16(const float4* __restrict__ src,
                                    us4* __restrict__ dst, int n4) {
    int i = blockIdx.x * blockDim.x + threadIdx.x;
    int stride = gridDim.x * blockDim.x;
    for (; i < n4; i += stride) {
        float4 v = src[i];
        us4 o;
        o.x = f32_to_bf16_rne(v.x);
        o.y = f32_to_bf16_rne(v.y);
        o.z = f32_to_bf16_rne(v.z);
        o.w = f32_to_bf16_rne(v.w);
        dst[i] = o;
    }
}

// ---------- row L2-normalization in place (bf16 rows) ----------
__global__ void rownorm_bf16(unsigned short* __restrict__ rows, int H) {
    unsigned short* p = rows + (size_t)blockIdx.x * H;
    float ss = 0.f;
    for (int i = threadIdx.x; i < H; i += blockDim.x) {
        float v = bf16_to_f32(p[i]);
        ss += v * v;
    }
    __shared__ float red[256];
    red[threadIdx.x] = ss;
    __syncthreads();
    for (int s = 128; s > 0; s >>= 1) {
        if ((int)threadIdx.x < s) red[threadIdx.x] += red[threadIdx.x + s];
        __syncthreads();
    }
    float scale = 1.0f / fmaxf(sqrtf(red[0]), 1e-8f);
    for (int i = threadIdx.x; i < H; i += blockDim.x) {
        p[i] = f32_to_bf16_rne(bf16_to_f32(p[i]) * scale);
    }
}

// ---------- main WMMA GEMM: C = A[M,K] * B[N,K]^T (+ epilogue) ----------
// Block: 128 threads = 4 waves (2x2), block tile 128x128, wave tile 64x64.
// Double-buffered LDS, async global->LDS pipeline.
// MODE 0: C_bf16 = relu(acc + bias[n])
// MODE 1: C_bf16 = acc + bias[n]
// MODE 2: C_f32  = (acc + 1) * 0.49
template<int MODE>
__global__ __launch_bounds__(128)
void gemm_bf16_wmma(const unsigned short* __restrict__ A,
                    const unsigned short* __restrict__ B,
                    const float* __restrict__ bias,
                    void* __restrict__ Cout,
                    int M, int N, int K) {
    constexpr int BM = 128, BN = 128, BK = 64;
    constexpr int LDT = BK + 8;                 // padded LDS row stride in halves
    constexpr int TILE_HALVES = BM * LDT;       // 9216 halves = 18432 B per matrix
    constexpr int TILE_BYTES  = TILE_HALVES * 2;

    extern __shared__ unsigned short smem[];
    // layout: [buf0: A | B][buf1: A | B]

    const int tid  = threadIdx.x;               // 0..127
    const int wave = tid >> 5;                  // 0..3
    const int lane = tid & 31;
    const int wrow = (wave >> 1) * 64;          // 2x2 wave grid, 64x64 wave tile
    const int wcol = (wave & 1) * 64;

    const int blockRow = blockIdx.y * BM;
    const int blockCol = blockIdx.x * BN;

    const int m16 = lane & 15;
    const int kh  = lane >> 4;

    v8f acc[4][4] = {};                         // 128 fp32 accum VGPRs

    // per-thread global/LDS mapping for async copies: thread t owns row t
    const unsigned short* gA = A + (size_t)(blockRow + tid) * K;
    const unsigned short* gB = B + (size_t)(blockCol + tid) * K;
    const uint32_t ldsBase = (uint32_t)(uintptr_t)(void*)smem;  // runtime conversion
    const uint32_t ldsRow  = ldsBase + (uint32_t)tid * (LDT * 2);

    auto issue_tile = [&](int k0, int buf) {
        uint32_t la = ldsRow + (uint32_t)buf * (2u * TILE_BYTES);
        uint32_t lb = la + TILE_BYTES;
#pragma unroll
        for (int c = 0; c < 8; c++)
            async_b128(la + c * 16, (uint64_t)(uintptr_t)(gA + k0 + c * 8));
#pragma unroll
        for (int c = 0; c < 8; c++)
            async_b128(lb + c * 16, (uint64_t)(uintptr_t)(gB + k0 + c * 8));
    };

    issue_tile(0, 0);

    int cur = 0;
    for (int k0 = 0; k0 < K; k0 += BK) {
        const bool hasNext = (k0 + BK) < K;
        if (hasNext) {
            issue_tile(k0 + BK, cur ^ 1);       // 16 asyncs in flight for next tile
            wait_async_le16();                  // current tile's 16 are complete
        } else {
            wait_async_0();
        }
        __syncthreads();

        const unsigned short* as = smem + cur * (2 * TILE_HALVES);
        const unsigned short* bs = as + TILE_HALVES;
#pragma unroll
        for (int ks = 0; ks < BK; ks += 32) {
            v16bf af[4], bf[4];
#pragma unroll
            for (int tm = 0; tm < 4; tm++) {
                // A 16x32 bf16: lanes 0-15 K {0..7,16..23}, lanes 16-31 {8..15,24..31}
                int r = wrow + tm * 16 + m16;
                uint4 q0 = *(const uint4*)&as[r * LDT + ks + kh * 8];
                uint4 q1 = *(const uint4*)&as[r * LDT + ks + 16 + kh * 8];
                af[tm] = pack_frag(q0, q1);
            }
#pragma unroll
            for (int tn = 0; tn < 4; tn++) {
                // B 32x16 bf16: lanes 0-15 K {0..7,16..23}, lanes 16-31 {8..15,24..31}
                int c = wcol + tn * 16 + m16;
                uint4 q0 = *(const uint4*)&bs[c * LDT + ks + kh * 8];
                uint4 q1 = *(const uint4*)&bs[c * LDT + ks + 16 + kh * 8];
                bf[tn] = pack_frag(q0, q1);
            }
#pragma unroll
            for (int tm = 0; tm < 4; tm++)
#pragma unroll
                for (int tn = 0; tn < 4; tn++)
                    acc[tm][tn] = __builtin_amdgcn_wmma_f32_16x16x32_bf16(
                        false, af[tm], false, bf[tn],
                        (short)0, acc[tm][tn], false, false);
        }
        __syncthreads();                        // done reading buf before it is refilled
        cur ^= 1;
    }

    // epilogue: D layout: VGPR i -> (M = i + 8*(lane>=16), N = lane&15)
#pragma unroll
    for (int tm = 0; tm < 4; tm++) {
#pragma unroll
        for (int tn = 0; tn < 4; tn++) {
            int colg = blockCol + wcol + tn * 16 + m16;
            float bv = (MODE == 2) ? 0.f : bias[colg];
#pragma unroll
            for (int i = 0; i < 8; i++) {
                int rowg = blockRow + wrow + tm * 16 + i + kh * 8;
                float v = acc[tm][tn][i] + bv;
                if (MODE == 0) v = fmaxf(v, 0.f);
                if (MODE == 2) {
                    ((float*)Cout)[(size_t)rowg * N + colg] = (v + 1.0f) * 0.49f;
                } else {
                    ((unsigned short*)Cout)[(size_t)rowg * N + colg] = f32_to_bf16_rne(v);
                }
            }
        }
    }
}

// ---------- launch ----------
extern "C" void kernel_launch(void* const* d_in, const int* in_sizes, int n_in,
                              void* d_out, int out_size, void* d_ws, size_t ws_size,
                              hipStream_t stream) {
    const int H = 1024, F = 8192, Kw = 4096;

    const float* feat    = (const float*)d_in[0];
    const float* keyword = (const float*)d_in[1];
    const float* W1 = (const float*)d_in[2];
    const float* b1 = (const float*)d_in[3];
    const float* W2 = (const float*)d_in[4];
    const float* b2 = (const float*)d_in[5];
    const float* Wk = (const float*)d_in[6];
    const float* bk = (const float*)d_in[7];
    float* out = (float*)d_out;

    char* ws = (char*)d_ws;
    unsigned short* featB = (unsigned short*)ws; ws += (size_t)F * H * 2;   // 16 MB
    unsigned short* keyB  = (unsigned short*)ws; ws += (size_t)Kw * H * 2;  //  8 MB
    unsigned short* W1B   = (unsigned short*)ws; ws += (size_t)H * H * 2;   //  2 MB
    unsigned short* W2B   = (unsigned short*)ws; ws += (size_t)H * H * 2;   //  2 MB
    unsigned short* WkB   = (unsigned short*)ws; ws += (size_t)H * H * 2;   //  2 MB
    unsigned short* h1B   = (unsigned short*)ws; ws += (size_t)F * H * 2;   // 16 MB
    unsigned short* fB    = (unsigned short*)ws; ws += (size_t)F * H * 2;   // 16 MB
    unsigned short* kB    = (unsigned short*)ws; ws += (size_t)Kw * H * 2;  //  8 MB

    auto cvt = [&](const float* s, unsigned short* d, size_t n) {
        int n4 = (int)(n / 4);
        int blocks = (n4 + 255) / 256;
        convert_f32_to_bf16<<<blocks, 256, 0, stream>>>((const float4*)s, (us4*)d, n4);
    };
    cvt(feat,    featB, (size_t)F * H);
    cvt(keyword, keyB,  (size_t)Kw * H);
    cvt(W1, W1B, (size_t)H * H);
    cvt(W2, W2B, (size_t)H * H);
    cvt(Wk, WkB, (size_t)H * H);

    // dynamic LDS: double-buffered A+B tiles = 4 * 128 * 72 halves = 73728 B
    const size_t smemBytes = 4u * 128u * 72u * 2u;

    // h1 = relu(feat @ W1^T + b1)    [F,H]
    gemm_bf16_wmma<0><<<dim3(H / 128, F / 128), 128, smemBytes, stream>>>(
        featB, W1B, b1, (void*)h1B, F, H, H);
    // f = h1 @ W2^T + b2             [F,H]
    gemm_bf16_wmma<1><<<dim3(H / 128, F / 128), 128, smemBytes, stream>>>(
        h1B, W2B, b2, (void*)fB, F, H, H);
    // k = keyword @ Wk^T + bk        [Kw,H]
    gemm_bf16_wmma<1><<<dim3(H / 128, Kw / 128), 128, smemBytes, stream>>>(
        keyB, WkB, bk, (void*)kB, Kw, H, H);

    // normalize rows of f and k
    rownorm_bf16<<<F,  256, 0, stream>>>(fB, H);
    rownorm_bf16<<<Kw, 256, 0, stream>>>(kB, H);

    // cos = k_hat @ f_hat^T ; out = (cos+1)*0.49   [Kw, F]
    gemm_bf16_wmma<2><<<dim3(F / 128, Kw / 128), 128, smemBytes, stream>>>(
        kB, fB, nullptr, (void*)out, Kw, F, H);
}
